// ESIM_4337916969354
// MI455X (gfx1250) — compile-verified, matrix operand order checked by reading, not run on hardware
//
#include <hip/hip_runtime.h>
#include <hip/hip_bf16.h>
#include <math.h>
#include <stdint.h>

// ---------------- problem dims ----------------
#define BQ    8      // query batch
#define SN    8      // paragraphs per query
#define QL    50     // query length
#define PL    200    // paragraph length
#define ED    300    // embed dim
#define EPAD  320    // embed dim padded to x32
#define HD    256    // LSTM hidden
#define H2    512    // bi hidden
#define G4    1024   // 4*H gates
#define NPAR  64     // B*S paragraph rows
#define D8H   2048   // 8*H combined feature
#define LINN  512
#define KT    32     // GEMM K-chunk

typedef __attribute__((ext_vector_type(16))) __bf16 v16bf;
typedef __attribute__((ext_vector_type(8)))  float  v8f;

// ---------------- device helpers ----------------
__device__ __forceinline__ uint16_t f2bf(float f) {
  union { float f; uint32_t u; } x; x.f = f;
  uint32_t r = x.u + 0x7FFFu + ((x.u >> 16) & 1u);   // round-to-nearest-even
  return (uint16_t)(r >> 16);
}

__device__ __forceinline__ float sigm(float x) { return 1.0f / (1.0f + expf(-x)); }

// Load a 16-lane-striped 16x32 bf16 fragment (A or B operand) per CDNA5 ISA
// layout: lane l holds row (l&15); k-chunks {base..base+7, base+16..base+23},
// base = 8*(l>=16).  Two 16-byte loads.
__device__ __forceinline__ v16bf load_frag(const uint16_t* base_p, int row, int ldk,
                                           int k0, int lane) {
  int half = (lane >> 4) & 1;
  const uint16_t* p = base_p + (size_t)row * ldk + k0 + half * 8;
  union { uint4 q[2]; v16bf v; } f;
  f.q[0] = *(const uint4*)(p);
  f.q[1] = *(const uint4*)(p + 16);
  return f.v;
}

// ---------------- conversion / gather kernels ----------------
__global__ void convw_k(const float* __restrict__ W, uint16_t* __restrict__ out,
                        int N, int K, int Kpad) {
  int idx = blockIdx.x * blockDim.x + threadIdx.x;
  int total = N * Kpad;
  if (idx >= total) return;
  int r = idx / Kpad, k = idx % Kpad;
  out[idx] = f2bf(k < K ? W[(size_t)r * K + k] : 0.0f);
}

// zero-fills both the K pad and rows in [nvalid, nalloc) so 32-row M blocking is safe
__global__ void gather_k(const int* __restrict__ tok, const float* __restrict__ emb,
                         uint16_t* __restrict__ out, int nvalid, int nalloc) {
  int idx = blockIdx.x * blockDim.x + threadIdx.x;
  int total = nalloc * EPAD;
  if (idx >= total) return;
  int r = idx / EPAD, k = idx % EPAD;
  float v = (r < nvalid && k < ED) ? emb[(size_t)tok[r] * ED + k] : 0.0f;
  out[idx] = f2bf(v);
}

// ---------------- WMMA GEMM:  C[MxN] = A[MxK]bf16 @ W[NxK]bf16^T + bias ----------------
// block = 128 threads = 4 waves.  Each wave owns a 32x32 output tile (2x2 WMMA block);
// the 4 waves share one 32-row A tile staged through LDS with double-buffered
// CDNA5 async global->LDS loads (ASYNCcnt).  B streams from global (L2-resident).
// Requires: N % 128 == 0, K % 32 == 0, A allocated with ceil(M/32)*32 rows.
__global__ __launch_bounds__(128) void gemm_bf16_nt(
    const uint16_t* __restrict__ A, const uint16_t* __restrict__ W,
    const float* __restrict__ bias, float* __restrict__ C,
    int M, int N, int K) {
  __shared__ __align__(16) uint16_t tile[2][32 * KT];
  int tid  = threadIdx.x;
  int wave = tid >> 5;
  int lane = tid & 31;
  int mBase = blockIdx.y * 32;
  int nBase = (blockIdx.x * 4 + wave) * 32;
  // staging: 32 rows x 32 bf16 = 2KB per chunk; 128 threads x 16B
  int sr = tid >> 2;                  // staging row 0..31
  int sc = (tid & 3) * 8;             // staging col (elements)
  const uint16_t* gA = A + (size_t)(mBase + sr) * K + sc;
  uint32_t lds0 = (uint32_t)(uintptr_t)(&tile[0][sr * KT + sc]);
  uint32_t lds1 = (uint32_t)(uintptr_t)(&tile[1][sr * KT + sc]);
  int KC = K / KT;
  {
    uint64_t ga = (uint64_t)(uintptr_t)gA;
    asm volatile("global_load_async_to_lds_b128 %0, %1, off"
                 :: "v"(lds0), "v"(ga) : "memory");
  }
  int br0 = nBase + (lane & 15);
  int br1 = br0 + 16;
  v8f acc00 = {}, acc01 = {}, acc10 = {}, acc11 = {};
  for (int kc = 0; kc < KC; ++kc) {
    int k0 = kc * KT;
    if (kc + 1 < KC) {
      // issue next A chunk into the other buffer, then wait for the current one
      uint64_t ga = (uint64_t)(uintptr_t)(gA + k0 + KT);
      uint32_t lb = (kc & 1) ? lds0 : lds1;
      asm volatile("global_load_async_to_lds_b128 %0, %1, off"
                   :: "v"(lb), "v"(ga) : "memory");
      __builtin_prefetch(W + (size_t)br0 * K + k0 + KT, 0, 3);
      asm volatile("s_wait_asynccnt 0x1" ::: "memory");
    } else {
      asm volatile("s_wait_asynccnt 0x0" ::: "memory");
    }
    __syncthreads();
    const uint16_t* tb = &tile[kc & 1][0];
    v16bf a0 = load_frag(tb, (lane & 15),      KT, 0, lane);
    v16bf a1 = load_frag(tb, 16 + (lane & 15), KT, 0, lane);
    v16bf b0 = load_frag(W, br0, K, k0, lane);
    v16bf b1 = load_frag(W, br1, K, k0, lane);
    acc00 = __builtin_amdgcn_wmma_f32_16x16x32_bf16(false, a0, false, b0, (short)0, acc00, false, false);
    acc01 = __builtin_amdgcn_wmma_f32_16x16x32_bf16(false, a0, false, b1, (short)0, acc01, false, false);
    acc10 = __builtin_amdgcn_wmma_f32_16x16x32_bf16(false, a1, false, b0, (short)0, acc10, false, false);
    acc11 = __builtin_amdgcn_wmma_f32_16x16x32_bf16(false, a1, false, b1, (short)0, acc11, false, false);
    __syncthreads();   // everyone done reading this buffer before it is restaged
  }
  int n0 = nBase + (lane & 15);
  float bv0 = bias ? bias[n0] : 0.0f;
  float bv1 = bias ? bias[n0 + 16] : 0.0f;
  int mr = (lane >> 4) << 3;
#pragma unroll
  for (int r = 0; r < 8; ++r) {
    int m0 = mBase + mr + r;
    int m1 = m0 + 16;
    if (m0 < M) {
      C[(size_t)m0 * N + n0]      = acc00[r] + bv0;
      C[(size_t)m0 * N + n0 + 16] = acc01[r] + bv1;
    }
    if (m1 < M) {
      C[(size_t)m1 * N + n0]      = acc10[r] + bv0;
      C[(size_t)m1 * N + n0 + 16] = acc11[r] + bv1;
    }
  }
}

// ---------------- LSTM recurrence (one dir, 16-row batch chunk per block) -----------
// Xproj[row, t, 1024] already holds x@Wih^T + b.  Per step: G = h_bf16 @ Whh^T (WMMA),
// gate math in-register, h/c state resident in LDS.  Backward direction handled by
// per-row time-index reversal (rev(t) = len-1-t inside the valid span).
__global__ __launch_bounds__(512) void lstm_rec(
    const float* __restrict__ Xproj, const uint16_t* __restrict__ Whh,
    const int* __restrict__ lens, int lenDiv,
    float* __restrict__ Out, int Ntot, int T, int dir) {
  __shared__ __align__(16) float    hS[16][HD];
  __shared__ __align__(16) float    cS[16][HD];
  __shared__ __align__(16) uint16_t hB[16][HD];
  int base = blockIdx.x * 16;
  for (int i = threadIdx.x; i < 16 * HD; i += blockDim.x) {
    (&hS[0][0])[i] = 0.0f; (&cS[0][0])[i] = 0.0f;
  }
  __syncthreads();
  int wave = threadIdx.x >> 5;      // 16 waves -> 16 n-tiles cover n = 0..255
  int lane = threadIdx.x & 31;
  int nl   = lane & 15;
  int n    = wave * 16 + nl;
  int arow = lane & 15;
  for (int t = 0; t < T; ++t) {
    for (int i = threadIdx.x; i < 16 * HD; i += blockDim.x)
      (&hB[0][0])[i] = f2bf((&hS[0][0])[i]);
    __syncthreads();
    v8f acc[4] = { v8f{}, v8f{}, v8f{}, v8f{} };
    for (int k0 = 0; k0 < HD; k0 += 32) {
      v16bf a = load_frag(&hB[0][0], arow, HD, k0, lane);
#pragma unroll
      for (int g = 0; g < 4; ++g) {
        v16bf b = load_frag(Whh, g * HD + n, HD, k0, lane);
        acc[g] = __builtin_amdgcn_wmma_f32_16x16x32_bf16(false, a, false, b,
                                                         (short)0, acc[g], false, false);
      }
    }
#pragma unroll
    for (int r = 0; r < 8; ++r) {
      int m = r + ((lane >> 4) << 3);     // 0..15
      int row = base + m;
      if (row < Ntot) {
        int len = lens[row / lenDiv];
        bool msk = t < len;
        int trev = msk ? (len - 1 - t) : t;
        int tix  = dir ? trev : t;
        const float* xp = Xproj + ((size_t)row * T + tix) * (size_t)G4;
        float gi = acc[0][r] + xp[n];
        float gf = acc[1][r] + xp[HD + n];
        float gg = acc[2][r] + xp[2 * HD + n];
        float go = acc[3][r] + xp[3 * HD + n];
        float i_ = sigm(gi), f_ = sigm(gf), g_ = tanhf(gg), o_ = sigm(go);
        float c2 = f_ * cS[m][n] + i_ * g_;
        float h2 = o_ * tanhf(c2);
        float ov = 0.0f;
        if (msk) { cS[m][n] = c2; hS[m][n] = h2; ov = h2; }
        Out[((size_t)row * T + tix) * (size_t)H2 + dir * HD + n] = ov;
      }
    }
    __syncthreads();
  }
}

// ---------------- attention (VALU; ~1% of FLOPs) ----------------
__global__ void att_k(const float* __restrict__ qo, const float* __restrict__ po,
                      float* __restrict__ att) {
  int bsq = blockIdx.x;                 // bs*QL + q
  int bs = bsq / QL, q = bsq % QL;
  int b = bs / SN;
  __shared__ float qv[H2];
  for (int i = threadIdx.x; i < H2; i += blockDim.x)
    qv[i] = qo[((size_t)b * QL + q) * H2 + i];
  __syncthreads();
  for (int p = threadIdx.x; p < PL; p += blockDim.x) {
    const float* pv = po + ((size_t)bs * PL + p) * H2;
    float s = 0.0f;
    for (int h = 0; h < H2; ++h) s += qv[h] * pv[h];
    att[(size_t)bsq * PL + p] = s;
  }
}

__global__ void p_align_k(const float* __restrict__ att, const int* __restrict__ qtok,
                          const float* __restrict__ qo, float* __restrict__ pal) {
  int bsp = blockIdx.x;                 // bs*PL + p
  int bs = bsp / PL, p = bsp % PL;
  int b = bs / SN;
  __shared__ float w[QL];
  for (int q = threadIdx.x; q < QL; q += blockDim.x)
    w[q] = (qtok[b * QL + q] == 0) ? -1000.0f : att[((size_t)bs * QL + q) * PL + p];
  __syncthreads();
  if (threadIdx.x == 0) {
    float mx = -3.4e38f;
    for (int q = 0; q < QL; ++q) mx = fmaxf(mx, w[q]);
    float sm = 0.0f;
    for (int q = 0; q < QL; ++q) { float e = expf(w[q] - mx); w[q] = e; sm += e; }
    float inv = 1.0f / sm;
    for (int q = 0; q < QL; ++q) w[q] *= inv;
  }
  __syncthreads();
  for (int h = threadIdx.x; h < H2; h += blockDim.x) {
    float s = 0.0f;
    for (int q = 0; q < QL; ++q) s += w[q] * qo[((size_t)b * QL + q) * H2 + h];
    pal[(size_t)bsp * H2 + h] = s;
  }
}

__global__ void q_align_k(const float* __restrict__ att, const int* __restrict__ ptok,
                          const float* __restrict__ po, float* __restrict__ qal) {
  int bsq = blockIdx.x;                 // bs*QL + q
  int bs = bsq / QL;
  __shared__ float w[PL];
  for (int p = threadIdx.x; p < PL; p += blockDim.x)
    w[p] = (ptok[bs * PL + p] == 0) ? -1000.0f : att[(size_t)bsq * PL + p];
  __syncthreads();
  if (threadIdx.x == 0) {
    float mx = -3.4e38f;
    for (int p = 0; p < PL; ++p) mx = fmaxf(mx, w[p]);
    float sm = 0.0f;
    for (int p = 0; p < PL; ++p) { float e = expf(w[p] - mx); w[p] = e; sm += e; }
    float inv = 1.0f / sm;
    for (int p = 0; p < PL; ++p) w[p] *= inv;
  }
  __syncthreads();
  for (int h = threadIdx.x; h < H2; h += blockDim.x) {
    float s = 0.0f;
    for (int p = 0; p < PL; ++p) s += w[p] * po[((size_t)bs * PL + p) * H2 + h];
    qal[(size_t)bsq * H2 + h] = s;
  }
}

// ---------------- combined features -> bf16 directly ----------------
__global__ void comb_k(const float* __restrict__ x, const float* __restrict__ a,
                       uint16_t* __restrict__ out, int nrows, int bcast) {
  int idx = blockIdx.x * blockDim.x + threadIdx.x;
  int total = nrows * D8H;
  if (idx >= total) return;
  int rr = idx / D8H, c = idx % D8H;
  int sel = c / H2, h = c % H2;
  int xr = bcast ? ((rr / (SN * QL)) * QL + (rr % QL)) : rr;
  float xv = x[(size_t)xr * H2 + h];
  float av = a[(size_t)rr * H2 + h];
  float v = (sel == 0) ? xv : (sel == 1) ? av : (sel == 2) ? xv * av : (xv - av);
  out[idx] = f2bf(v);
}

// ---------------- pooling + MLP head ----------------
__global__ void pool_k(const float* __restrict__ pco, const float* __restrict__ qco,
                       float* __restrict__ feat) {
  int row = blockIdx.x;
  for (int h = threadIdx.x; h < H2; h += blockDim.x) {
    float sm = 0.0f, mx = -3.4e38f;
    for (int t = 0; t < PL; ++t) {
      float v = pco[((size_t)row * PL + t) * H2 + h]; sm += v; mx = fmaxf(mx, v);
    }
    feat[(size_t)row * D8H + h]      = sm / (float)PL;
    feat[(size_t)row * D8H + H2 + h] = mx;
    sm = 0.0f; mx = -3.4e38f;
    for (int t = 0; t < QL; ++t) {
      float v = qco[((size_t)row * QL + t) * H2 + h]; sm += v; mx = fmaxf(mx, v);
    }
    feat[(size_t)row * D8H + 2 * H2 + h] = sm / (float)QL;
    feat[(size_t)row * D8H + 3 * H2 + h] = mx;
  }
}

__global__ void bn_k(const float* __restrict__ x, const float* __restrict__ g,
                     const float* __restrict__ b, const float* __restrict__ m,
                     const float* __restrict__ v, float* __restrict__ y,
                     int rows, int C) {
  int idx = blockIdx.x * blockDim.x + threadIdx.x;
  int total = rows * C;
  if (idx >= total) return;
  int c = idx % C;
  y[idx] = g[c] * (x[idx] - m[c]) * rsqrtf(v[c] + 1e-5f) + b[c];
}

__global__ void linear_k(const float* __restrict__ x, const float* __restrict__ W,
                         const float* __restrict__ b, float* __restrict__ y,
                         int N, int K, int elu) {
  int r = blockIdx.x;
  const float* xr = x + (size_t)r * K;
  for (int j = threadIdx.x; j < N; j += blockDim.x) {
    const float* wr = W + (size_t)j * K;
    float s = b[j];
    for (int k = 0; k < K; ++k) s += xr[k] * wr[k];
    if (elu) s = (s > 0.0f) ? s : (expf(s) - 1.0f);
    y[(size_t)r * N + j] = s;
  }
}

__global__ void final_k(const float* __restrict__ x, const float* __restrict__ W3,
                        const float* __restrict__ b3, float* __restrict__ out,
                        int rows, int K) {
  int r = blockIdx.x * blockDim.x + threadIdx.x;
  if (r >= rows) return;
  float s = b3[0];
  for (int k = 0; k < K; ++k) s += x[(size_t)r * K + k] * W3[k];
  out[r] = s;
}

// ---------------- host orchestration ----------------
extern "C" void kernel_launch(void* const* d_in, const int* in_sizes, int n_in,
                              void* d_out, int out_size, void* d_ws, size_t ws_size,
                              hipStream_t stream) {
  (void)in_sizes; (void)n_in; (void)out_size; (void)ws_size;
  const int*   d_query = (const int*)d_in[0];
  const int*   d_qlen  = (const int*)d_in[1];
  const int*   d_para  = (const int*)d_in[2];
  const int*   d_plen  = (const int*)d_in[3];
  const float* d_embed = (const float*)d_in[4];
  const float* qe[6]; const float* pe[6]; const float* iq[6]; const float* ip[6];
  for (int i = 0; i < 6; ++i) {
    qe[i] = (const float*)d_in[5 + i];
    pe[i] = (const float*)d_in[11 + i];
    iq[i] = (const float*)d_in[17 + i];
    ip[i] = (const float*)d_in[23 + i];
  }
  const float* bn1g = (const float*)d_in[29]; const float* bn1b = (const float*)d_in[30];
  const float* bn1m = (const float*)d_in[31]; const float* bn1v = (const float*)d_in[32];
  const float* W1 = (const float*)d_in[33];   const float* b1 = (const float*)d_in[34];
  const float* bn2g = (const float*)d_in[35]; const float* bn2b = (const float*)d_in[36];
  const float* bn2m = (const float*)d_in[37]; const float* bn2v = (const float*)d_in[38];
  const float* W2 = (const float*)d_in[39];   const float* b2 = (const float*)d_in[40];
  const float* bn3g = (const float*)d_in[41]; const float* bn3b = (const float*)d_in[42];
  const float* bn3m = (const float*)d_in[43]; const float* bn3v = (const float*)d_in[44];
  const float* W3 = (const float*)d_in[45];   const float* b3 = (const float*)d_in[46];

  char* ws = (char*)d_ws;
  size_t off = 0;
  auto alloc = [&](size_t bytes) -> void* {
    void* p = ws + off; off += (bytes + 255) & ~(size_t)255; return p;
  };
  const int QROWS  = BQ * QL;                 // 400 valid
  const int QROWSA = ((QROWS + 31) / 32) * 32; // 416 allocated (32-row M blocks)
  // bf16 weights
  uint16_t* wq_f  = (uint16_t*)alloc((size_t)G4 * EPAD * 2);
  uint16_t* wq_b  = (uint16_t*)alloc((size_t)G4 * EPAD * 2);
  uint16_t* wp_f  = (uint16_t*)alloc((size_t)G4 * EPAD * 2);
  uint16_t* wp_b  = (uint16_t*)alloc((size_t)G4 * EPAD * 2);
  uint16_t* wiq_f = (uint16_t*)alloc((size_t)G4 * D8H * 2);
  uint16_t* wiq_b = (uint16_t*)alloc((size_t)G4 * D8H * 2);
  uint16_t* wip_f = (uint16_t*)alloc((size_t)G4 * D8H * 2);
  uint16_t* wip_b = (uint16_t*)alloc((size_t)G4 * D8H * 2);
  uint16_t* hh[8];
  for (int i = 0; i < 8; ++i) hh[i] = (uint16_t*)alloc((size_t)G4 * HD * 2);
  // activations
  uint16_t* qemb = (uint16_t*)alloc((size_t)QROWSA * EPAD * 2);
  uint16_t* pemb = (uint16_t*)alloc((size_t)NPAR * PL * EPAD * 2);
  float* XA = (float*)alloc((size_t)NPAR * PL * G4 * 4);   // reused: p_enc f -> infer_p f
  float* XB = (float*)alloc((size_t)NPAR * PL * G4 * 4);   // reused: p_enc b -> infer_p b
  float* XC = (float*)alloc((size_t)NPAR * QL * G4 * 4);   // reused: q_enc f -> infer_q f
  float* XD = (float*)alloc((size_t)NPAR * QL * G4 * 4);   // reused: q_enc b -> infer_q b
  float* q_out = (float*)alloc((size_t)BQ * QL * H2 * 4);
  float* p_out = (float*)alloc((size_t)NPAR * PL * H2 * 4);
  float* att   = (float*)alloc((size_t)NPAR * QL * PL * 4);
  float* pal   = (float*)alloc((size_t)NPAR * PL * H2 * 4);
  float* qal   = (float*)alloc((size_t)NPAR * QL * H2 * 4);
  uint16_t* pcomb = (uint16_t*)alloc((size_t)NPAR * PL * D8H * 2);
  uint16_t* qcomb = (uint16_t*)alloc((size_t)NPAR * QL * D8H * 2);
  float* p_co = (float*)alloc((size_t)NPAR * PL * H2 * 4);
  float* q_co = (float*)alloc((size_t)NPAR * QL * H2 * 4);
  float* feat = (float*)alloc((size_t)NPAR * D8H * 4);
  float* t0   = (float*)alloc((size_t)NPAR * D8H * 4);
  float* t1   = (float*)alloc((size_t)NPAR * LINN * 4);
  float* t2   = (float*)alloc((size_t)NPAR * LINN * 4);

  auto gs = [](int total) { return dim3((unsigned)((total + 255) / 256)); };
  auto conv = [&](const float* W, uint16_t* o, int N, int K, int Kp) {
    convw_k<<<gs(N * Kp), 256, 0, stream>>>(W, o, N, K, Kp);
  };
  auto gemm = [&](const uint16_t* A, const uint16_t* W, const float* bias, float* C,
                  int M, int N, int K) {
    dim3 g((unsigned)(N / 128), (unsigned)((M + 31) / 32));
    gemm_bf16_nt<<<g, 128, 0, stream>>>(A, W, bias, C, M, N, K);
  };

  // 0) weight conversion to bf16 (padded K)
  conv(qe[0], wq_f, G4, ED, EPAD);  conv(qe[3], wq_b, G4, ED, EPAD);
  conv(pe[0], wp_f, G4, ED, EPAD);  conv(pe[3], wp_b, G4, ED, EPAD);
  conv(iq[0], wiq_f, G4, D8H, D8H); conv(iq[3], wiq_b, G4, D8H, D8H);
  conv(ip[0], wip_f, G4, D8H, D8H); conv(ip[3], wip_b, G4, D8H, D8H);
  conv(qe[1], hh[0], G4, HD, HD);   conv(qe[4], hh[1], G4, HD, HD);
  conv(pe[1], hh[2], G4, HD, HD);   conv(pe[4], hh[3], G4, HD, HD);
  conv(iq[1], hh[4], G4, HD, HD);   conv(iq[4], hh[5], G4, HD, HD);
  conv(ip[1], hh[6], G4, HD, HD);   conv(ip[4], hh[7], G4, HD, HD);

  // 1) embedding gather -> bf16 (zero-filled pad rows for 32-row M blocks)
  gather_k<<<gs(QROWSA * EPAD), 256, 0, stream>>>(d_query, d_embed, qemb, QROWS, QROWSA);
  gather_k<<<gs(NPAR * PL * EPAD), 256, 0, stream>>>(d_para, d_embed, pemb,
                                                     NPAR * PL, NPAR * PL);

  // 2) encoder input projections (WMMA + async-LDS A staging)
  gemm(qemb, wq_f, qe[2], XC, QROWS, G4, EPAD);
  gemm(qemb, wq_b, qe[5], XD, QROWS, G4, EPAD);
  gemm(pemb, wp_f, pe[2], XA, NPAR * PL, G4, EPAD);
  gemm(pemb, wp_b, pe[5], XB, NPAR * PL, G4, EPAD);

  // 3) encoder recurrences (WMMA h@Whh^T, LDS-resident state)
  lstm_rec<<<1, 512, 0, stream>>>(XC, hh[0], d_qlen, 1, q_out, BQ, QL, 0);
  lstm_rec<<<1, 512, 0, stream>>>(XD, hh[1], d_qlen, 1, q_out, BQ, QL, 1);
  lstm_rec<<<4, 512, 0, stream>>>(XA, hh[2], d_plen, 1, p_out, NPAR, PL, 0);
  lstm_rec<<<4, 512, 0, stream>>>(XB, hh[3], d_plen, 1, p_out, NPAR, PL, 1);

  // 4) co-attention (tiny FLOPs -> VALU)
  att_k<<<NPAR * QL, 256, 0, stream>>>(q_out, p_out, att);
  p_align_k<<<NPAR * PL, 256, 0, stream>>>(att, d_query, q_out, pal);
  q_align_k<<<NPAR * QL, 256, 0, stream>>>(att, d_para, p_out, qal);

  // 5) combined features directly in bf16
  comb_k<<<gs(NPAR * PL * D8H), 256, 0, stream>>>(p_out, pal, pcomb, NPAR * PL, 0);
  comb_k<<<gs(NPAR * QL * D8H), 256, 0, stream>>>(q_out, qal, qcomb, NPAR * QL, 1);

  // 6) inference input projections (dominant GEMMs, WMMA) -- reuse X arenas
  gemm(pcomb, wip_f, ip[2], XA, NPAR * PL, G4, D8H);
  gemm(pcomb, wip_b, ip[5], XB, NPAR * PL, G4, D8H);
  gemm(qcomb, wiq_f, iq[2], XC, NPAR * QL, G4, D8H);
  gemm(qcomb, wiq_b, iq[5], XD, NPAR * QL, G4, D8H);

  // 7) inference recurrences
  lstm_rec<<<4, 512, 0, stream>>>(XA, hh[6], d_plen, 1, p_co, NPAR, PL, 0);
  lstm_rec<<<4, 512, 0, stream>>>(XB, hh[7], d_plen, 1, p_co, NPAR, PL, 1);
  lstm_rec<<<4, 512, 0, stream>>>(XC, hh[4], d_qlen, SN, q_co, NPAR, QL, 0);
  lstm_rec<<<4, 512, 0, stream>>>(XD, hh[5], d_qlen, SN, q_co, NPAR, QL, 1);

  // 8) pooling + MLP head
  pool_k<<<NPAR, 256, 0, stream>>>(p_co, q_co, feat);
  bn_k<<<gs(NPAR * D8H), 256, 0, stream>>>(feat, bn1g, bn1b, bn1m, bn1v, t0, NPAR, D8H);
  linear_k<<<NPAR, 256, 0, stream>>>(t0, W1, b1, t1, LINN, D8H, 1);
  bn_k<<<gs(NPAR * LINN), 256, 0, stream>>>(t1, bn2g, bn2b, bn2m, bn2v, t2, NPAR, LINN);
  linear_k<<<NPAR, 256, 0, stream>>>(t2, W2, b2, t1, LINN, LINN, 1);
  bn_k<<<gs(NPAR * LINN), 256, 0, stream>>>(t1, bn3g, bn3b, bn3m, bn3v, t2, NPAR, LINN);
  final_k<<<1, 64, 0, stream>>>(t2, W3, b3, (float*)d_out, NPAR, LINN);
}
